// MultiheadLshAttention_47253230191326
// MI455X (gfx1250) — matrix operand order repaired and checked by reading, not compiled, not run on hardware
//
#include <hip/hip_runtime.h>
#include <cstdint>

// ---------------------------------------------------------------------------
// MultiheadLshAttention for MI455X (gfx1250), wave32 + WMMA bf16 + TDM.
// T=4096 B=2 E=1024 H=16 D=64 R=2 NH=16 CHUNK=128 -> C=32, pad=0.
// share_kq: hash(q)==hash(k) (positive per-row scaling), so one sort.
// ---------------------------------------------------------------------------

typedef __bf16 bf16;
typedef uint32_t u32;
typedef __attribute__((ext_vector_type(16))) __bf16   v16bf;
typedef __attribute__((ext_vector_type(8)))  float    v8f;
typedef __attribute__((ext_vector_type(4)))  uint32_t v4u;
typedef __attribute__((ext_vector_type(8)))  uint32_t v8u;

constexpr int Tt   = 4096;
constexpr int Bb   = 2;
constexpr int Ee   = 1024;
constexpr int Hh   = 16;
constexpr int Dd   = 64;
constexpr int Rr   = 2;
constexpr int CSz  = 128;
constexpr int Cc   = Tt / CSz;          // 32
constexpr int Nrows = Tt * Bb;          // 8192
constexpr int NBRH  = Bb * Rr * Hh;     // 64

// ---------------- WMMA helpers (contiguous k order: lane half h holds k 16h..16h+15) ----------------

__device__ __forceinline__ v8f wmma_bf16(v16bf a, v16bf b, v8f c) {
  return __builtin_amdgcn_wmma_f32_16x16x32_bf16(false, a, false, b,
                                                 (short)0, c, false, false);
}

// A fragment 16x32 (bf16), storage row-major [m][k], lds stride ldk.
// lanes 0-15: M=0..15, K 0..15; lanes 16-31: K 16..31.
__device__ __forceinline__ v16bf load_a_frag(const bf16* p, int ldk) {
  int lane = threadIdx.x & 31;
  int m = lane & 15, kh = (lane >> 4) * 16;
  const bf16* row = p + m * ldk;
  v16bf f;
#pragma unroll
  for (int e = 0; e < 16; ++e) f[e] = row[kh + e];
  return f;
}

// B fragment 32x16 (bf16) with storage [col][k] (column-major wrt B).
__device__ __forceinline__ v16bf load_b_colmajor(const bf16* p, int ldk) {
  int lane = threadIdx.x & 31;
  int cc = lane & 15, kh = (lane >> 4) * 16;
  const bf16* row = p + cc * ldk;
  v16bf f;
#pragma unroll
  for (int e = 0; e < 16; ++e) f[e] = row[kh + e];
  return f;
}

// B fragment 32x16 (bf16) with storage [k][col] (row-major wrt B), stride ld.
__device__ __forceinline__ v16bf load_b_rowmajor(const bf16* p, int ld) {
  int lane = threadIdx.x & 31;
  int cc = lane & 15, kh = (lane >> 4) * 16;
  v16bf f;
#pragma unroll
  for (int e = 0; e < 16; ++e) f[e] = p[(kh + e) * ld + cc];
  return f;
}

// ---------------- TDM: 2D tile (128 rows x 64 bf16) global -> LDS ------------
// D# per CDNA5 ISA. LDS dest padded to 68 elems
// per row via pad_enable (interval = 32 DWORDs = one row, amount = 2 DWORDs).

__device__ __forceinline__ void tdm_load_tile_128x64(const void* gptr,
                                                     void* lptr) {
  u32 lds = (u32)(uintptr_t)lptr;                 // low 32 bits = LDS offset
  uint64_t ga = (uint64_t)(uintptr_t)gptr;        // 57-bit global byte addr
  v4u g0;
  g0[0] = 1u;                                     // count=1, user mode
  g0[1] = lds;                                    // lds_addr
  g0[2] = (u32)ga;                                // global_addr[31:0]
  g0[3] = (u32)(ga >> 32) | (2u << 30);           // global_addr[56:32], type=2
  v8u g1;
  g1[0] = (1u << 16)     // data_size = 1 (2 bytes)
        | (1u << 20)     // pad_enable
        | (4u << 22)     // pad_interval: 32 DWORDs (one 128B row)
        | (1u << 25);    // pad_amount: 2 DWORDs (8B -> stride 68 elems)
  g1[1] = 64u << 16;     // tensor_dim0 = 64 (bits 79:48)
  g1[2] = 128u << 16;    // tensor_dim1 = 128 (bits 111:80)
  g1[3] = 64u << 16;     // tile_dim0 = 64 (bits 127:112)
  g1[4] = 128u;          // tile_dim1 = 128 (bits 143:128), tile_dim2 = 0
  g1[5] = 64u;           // tensor_dim0_stride = 64 (bits 207:160)
  g1[6] = 0u;
  g1[7] = 0u;            // tensor_dim1_stride = 0 (unused, 2D)
  asm volatile("tensor_load_to_lds %0, %1" ::"s"(g0), "s"(g1) : "memory");
}

// ---------------- GEMM: C[Nr][Nc] = A[Nr][K] @ W[Nc][K]^T + bias -------------
// Block 128x128, BK=32, 8 waves: wave = 32 rows x 64 cols (2x4 tiles).

template <bool ABF16>
__global__ __launch_bounds__(256) void gemm_xwT_kernel(
    const void* __restrict__ A_, const float* __restrict__ W,
    const float* __restrict__ bias, float* __restrict__ outF,
    bf16* __restrict__ outB, int Nr, int K, int Nc) {
  constexpr int LDK = 40;  // 32 + pad
  __shared__ bf16 As[128 * LDK];
  __shared__ bf16 Bs[128 * LDK];

  const int bn = blockIdx.x * 128;  // output col base
  const int bm = blockIdx.y * 128;  // output row base
  const int tid = threadIdx.x;
  const int w = tid >> 5, lane = tid & 31;
  const int wm = (w >> 1) * 32, wn = (w & 1) * 64;

  v8f acc[2][4] = {};

  for (int k0 = 0; k0 < K; k0 += 32) {
    for (int i = tid; i < 128 * 32; i += 256) {
      int r = i >> 5, cc = i & 31;
      bf16 av;
      if (ABF16)
        av = ((const bf16*)A_)[(size_t)(bm + r) * K + k0 + cc];
      else
        av = (bf16)(((const float*)A_)[(size_t)(bm + r) * K + k0 + cc]);
      As[r * LDK + cc] = av;
      Bs[r * LDK + cc] = (bf16)(W[(size_t)(bn + r) * K + k0 + cc]);
    }
    __syncthreads();

    v16bf afr[2], bfr[4];
#pragma unroll
    for (int rt = 0; rt < 2; ++rt)
      afr[rt] = load_a_frag(As + (wm + rt * 16) * LDK, LDK);
#pragma unroll
    for (int ct = 0; ct < 4; ++ct)
      bfr[ct] = load_b_colmajor(Bs + (wn + ct * 16) * LDK, LDK);
#pragma unroll
    for (int rt = 0; rt < 2; ++rt)
#pragma unroll
      for (int ct = 0; ct < 4; ++ct)
        acc[rt][ct] = wmma_bf16(afr[rt], bfr[ct], acc[rt][ct]);
    __syncthreads();
  }

  const int hf = lane >> 4, ln = lane & 15;
#pragma unroll
  for (int rt = 0; rt < 2; ++rt)
#pragma unroll
    for (int ct = 0; ct < 4; ++ct)
#pragma unroll
      for (int r = 0; r < 8; ++r) {
        int row = bm + wm + rt * 16 + r + 8 * hf;
        int col = bn + wn + ct * 16 + ln;
        float v = acc[rt][ct][r] + bias[col];
        if (outF) outF[(size_t)row * Nc + col] = v;
        if (outB) outB[(size_t)row * Nc + col] = (bf16)v;
      }
}

// ---------------- row L2 norm: qbf = q*D^-0.5, kbf = q/||q|| -----------------

__global__ __launch_bounds__(256) void norm_scale_kernel(
    const float* __restrict__ q, bf16* __restrict__ qbf,
    bf16* __restrict__ kbf) {
  __shared__ float red[256];
  const int n = blockIdx.x;
  const float* row = q + (size_t)n * Ee;
  float s = 0.f;
  for (int e = threadIdx.x; e < Ee; e += 256) {
    float x = row[e];
    s += x * x;
  }
  red[threadIdx.x] = s;
  __syncthreads();
  for (int st = 128; st > 0; st >>= 1) {
    if (threadIdx.x < st) red[threadIdx.x] += red[threadIdx.x + st];
    __syncthreads();
  }
  const float rn = rsqrtf(red[0] + 1e-30f);
  for (int e = threadIdx.x; e < Ee; e += 256) {
    float x = row[e];
    qbf[(size_t)n * Ee + e] = (bf16)(x * 0.125f);  // D^-0.5 = 1/8
    kbf[(size_t)n * Ee + e] = (bf16)(x * rn);
  }
}

// ---------------- hash: argmax over [lin, -lin], 8 projections ---------------

__global__ __launch_bounds__(256) void hash_kernel(
    const float* __restrict__ q, const float* __restrict__ hw,
    int* __restrict__ buckets) {
  int gid = blockIdx.x * 256 + threadIdx.x;
  if (gid >= Nrows * Rr * Hh) return;
  int n = gid / (Rr * Hh), rh = gid % (Rr * Hh);
  int r = rh / Hh, h = rh % Hh;
  int t = n / Bb, b = n % Bb;
  const float* x = q + (size_t)n * Ee + h * Dd;
  const float* wp = hw + ((size_t)(r * Hh + h) * Dd) * 8;
  float acc[8] = {};
  for (int d = 0; d < Dd; ++d) {
    float xv = x[d];
#pragma unroll
    for (int j = 0; j < 8; ++j) acc[j] += xv * wp[d * 8 + j];
  }
  int best = 0;
  float bv = acc[0];
#pragma unroll
  for (int j = 1; j < 8; ++j)
    if (acc[j] > bv) { bv = acc[j]; best = j; }
#pragma unroll
  for (int j = 0; j < 8; ++j)
    if (-acc[j] > bv) { bv = -acc[j]; best = 8 + j; }
  buckets[(size_t)((b * Rr + r) * Hh + h) * Tt + t] = best;
}

// ---------------- stable counting sort per (b,r,h): 4096 keys, 16 buckets ----

__global__ __launch_bounds__(256) void sort_kernel(
    const int* __restrict__ buckets, int* __restrict__ idx,
    int* __restrict__ invp, int* __restrict__ sbuck) {
  __shared__ int cnt[16][256];
  __shared__ int totals[16];
  __shared__ int starts[16];
  const int brh = blockIdx.x;
  const int tid = threadIdx.x;
  const int* bk = buckets + (size_t)brh * Tt;
  const int base = tid * 16;
  int mybk[16], loc[16];
#pragma unroll
  for (int v = 0; v < 16; ++v) loc[v] = 0;
#pragma unroll
  for (int i = 0; i < 16; ++i) {
    int v = bk[base + i];
    mybk[i] = v;
    loc[v]++;
  }
#pragma unroll
  for (int v = 0; v < 16; ++v) cnt[v][tid] = loc[v];
  __syncthreads();
  if (tid < 16) {  // exclusive scan across threads for bucket `tid`
    int s = 0;
    for (int j = 0; j < 256; ++j) {
      int c0 = cnt[tid][j];
      cnt[tid][j] = s;
      s += c0;
    }
    totals[tid] = s;
  }
  __syncthreads();
  if (tid == 0) {
    int s = 0;
    for (int v = 0; v < 16; ++v) {
      starts[v] = s;
      s += totals[v];
    }
  }
  __syncthreads();
  int off[16];
#pragma unroll
  for (int v = 0; v < 16; ++v) off[v] = starts[v] + cnt[v][tid];
#pragma unroll
  for (int i = 0; i < 16; ++i) {
    int v = mybk[i];
    int p = off[v]++;
    int t = base + i;
    idx[(size_t)brh * Tt + p] = t;
    invp[(size_t)brh * Tt + t] = p;
    sbuck[(size_t)brh * Tt + p] = v;
  }
}

// ---------------- gather sorted q/k/v head slices (bf16) ---------------------

__global__ __launch_bounds__(256) void gather_kernel(
    const bf16* __restrict__ qbf, const bf16* __restrict__ kbf,
    const bf16* __restrict__ vbf, const int* __restrict__ idx,
    bf16* __restrict__ qsrt, bf16* __restrict__ ksrt, bf16* __restrict__ vsrt) {
  size_t gid = (size_t)blockIdx.x * 256 + threadIdx.x;
  int row = (int)(gid >> 6);
  int d = (int)(gid & 63);
  if (row >= NBRH * Tt) return;
  int brh = row / Tt;
  int h = brh % Hh;
  int br = brh / Hh;
  int b = br / Rr;
  int t = idx[row];
  int n = t * Bb + b;
  size_t src = (size_t)n * Ee + h * Dd + d;
  size_t dst = (size_t)row * Dd + d;
  qsrt[dst] = qbf[src];
  ksrt[dst] = kbf[src];
  vsrt[dst] = vbf[src];
}

// ---------------- flash-style chunked LSH attention --------------------------
// grid.x = NBRH*C. Block: 8 waves; wave owns 16 query rows. Key window = the
// up-to-3 neighbor chunks; clamped-duplicate chunks are skipped entirely,
// which exactly matches the reference's first-occurrence `off_ok` mask.
// One online-softmax update per 128-key chunk; K/V/Q tiles staged by the TDM.

__global__ __launch_bounds__(256) void lsh_attn_kernel(
    const bf16* __restrict__ qsrt, const bf16* __restrict__ ksrt,
    const bf16* __restrict__ vsrt, const int* __restrict__ sbuck,
    float* __restrict__ out_s, float* __restrict__ lse_s) {
  constexpr int LDQ = 68;  // 64 + pad (matches TDM pad_amount)
  __shared__ bf16 Qs[128 * LDQ];
  __shared__ bf16 Ks[128 * LDQ];
  __shared__ bf16 Vs[128 * LDQ];
  __shared__ bf16 Ps[8][16 * 36];  // per-wave P staging, 16x32 pad to 36
  __shared__ int qb[128];
  __shared__ int kb[128];

  const int blk = blockIdx.x;
  const int c = blk % Cc;
  const int brh = blk / Cc;
  const size_t base = (size_t)brh * Tt * Dd;
  const int tid = threadIdx.x, w = tid >> 5, lane = tid & 31;
  const int hf = lane >> 4, ln = lane & 15;

  // Q tile via TDM (wave 0 issues, waits, barrier publishes).
  if (w == 0) {
    tdm_load_tile_128x64(qsrt + base + (size_t)(c * CSz) * Dd, Qs);
    __builtin_amdgcn_s_wait_tensorcnt(0);
  }
  if (tid < 128) qb[tid] = sbuck[(size_t)brh * Tt + c * CSz + tid];
  __syncthreads();

  int myqb[8];
#pragma unroll
  for (int r = 0; r < 8; ++r) myqb[r] = qb[w * 16 + r + 8 * hf];
  const v16bf aq0 = load_a_frag(Qs + (w * 16) * LDQ, LDQ);       // d 0..31
  const v16bf aq1 = load_a_frag(Qs + (w * 16) * LDQ + 32, LDQ);  // d 32..63

  float m_r[8], l_r[8];
  v8f O[4] = {};
#pragma unroll
  for (int r = 0; r < 8; ++r) { m_r[r] = -1e30f; l_r[r] = 0.f; }

  for (int off = -1; off <= 1; ++off) {
    if (off == 0 && c == 0) continue;        // dup of clamped -1 chunk
    if (off == 1 && c == Cc - 1) continue;   // dup of center chunk
    int kc = c + off;
    if (kc < 0) kc = 0;
    if (kc >= Cc) kc = Cc - 1;

    __syncthreads();  // previous K/V consumption done
    if (w == 0) {
      const size_t g = base + (size_t)(kc * CSz) * Dd;
      tdm_load_tile_128x64(ksrt + g, Ks);
      tdm_load_tile_128x64(vsrt + g, Vs);
      __builtin_amdgcn_s_wait_tensorcnt(0);
    }
    if (tid < 128) kb[tid] = sbuck[(size_t)brh * Tt + kc * CSz + tid];
    __syncthreads();

    // ---- all 8 S tiles for this 128-key chunk (16 back-to-back WMMAs) ----
    v8f s[8];
#pragma unroll
    for (int tt = 0; tt < 8; ++tt) {
      v16bf b0 = load_b_colmajor(Ks + (tt * 16) * LDQ, LDQ);
      v16bf b1 = load_b_colmajor(Ks + (tt * 16) * LDQ + 32, LDQ);
      v8f z = {};
      z = wmma_bf16(aq0, b0, z);
      s[tt] = wmma_bf16(aq1, b1, z);
    }
    int kbl[8];
#pragma unroll
    for (int tt = 0; tt < 8; ++tt) kbl[tt] = kb[tt * 16 + ln];

    // ---- single online-softmax update over 128 keys ----
#pragma unroll
    for (int r = 0; r < 8; ++r) {
      float mx = -1e30f;
#pragma unroll
      for (int tt = 0; tt < 8; ++tt)
        if (kbl[tt] == myqb[r]) mx = fmaxf(mx, s[tt][r]);
#pragma unroll
      for (int dsh = 1; dsh < 16; dsh <<= 1)
        mx = fmaxf(mx, __shfl_xor(mx, dsh, 32));
      float mnew = fmaxf(m_r[r], mx);
      float scale = __expf(m_r[r] - mnew);
      float sm = 0.f;
#pragma unroll
      for (int tt = 0; tt < 8; ++tt) {
        float e = (kbl[tt] == myqb[r]) ? __expf(s[tt][r] - mnew) : 0.f;
        s[tt][r] = e;  // reuse S registers as P
        sm += e;
      }
#pragma unroll
      for (int dsh = 1; dsh < 16; dsh <<= 1) sm += __shfl_xor(sm, dsh, 32);
      l_r[r] = l_r[r] * scale + sm;
      m_r[r] = mnew;
#pragma unroll
      for (int ct = 0; ct < 4; ++ct) O[ct][r] *= scale;
    }

    // ---- P @ V in 4 groups of 32 keys (C-layout -> A-layout via LDS) ----
    bf16* pw = &Ps[w][0];
#pragma unroll
    for (int g = 0; g < 4; ++g) {
#pragma unroll
      for (int r = 0; r < 8; ++r) {
        int M = r + 8 * hf;
        pw[M * 36 + ln] = (bf16)s[2 * g][r];
        pw[M * 36 + 16 + ln] = (bf16)s[2 * g + 1][r];
      }
      asm volatile("s_wait_dscnt 0" ::: "memory");
      v16bf ap = load_a_frag(pw, 36);
#pragma unroll
      for (int ct = 0; ct < 4; ++ct) {
        v16bf bv = load_b_rowmajor(Vs + (g * 32) * LDQ + ct * 16, LDQ);
        O[ct] = wmma_bf16(ap, bv, O[ct]);
      }
    }
  }

#pragma unroll
  for (int r = 0; r < 8; ++r) {
    float il = 1.f / fmaxf(l_r[r], 1e-30f);
    int trow = c * CSz + w * 16 + r + 8 * hf;
#pragma unroll
    for (int ct = 0; ct < 4; ++ct)
      out_s[base + (size_t)trow * Dd + ct * 16 + ln] = O[ct][r] * il;
    if (ln == 0)
      lse_s[(size_t)brh * Tt + trow] = m_r[r] + __logf(fmaxf(l_r[r], 1e-30f));
  }
}

// ---------------- un-sort + combine rounds (softmax over LSE) ----------------

__global__ __launch_bounds__(256) void combine_kernel(
    const float* __restrict__ out_s, const float* __restrict__ lse_s,
    const int* __restrict__ invp, bf16* __restrict__ ctx) {
  size_t gid = (size_t)blockIdx.x * 256 + threadIdx.x;
  if (gid >= (size_t)Nrows * Ee) return;
  int n = (int)(gid / Ee), e = (int)(gid % Ee);
  int h = e >> 6, d = e & 63;
  int t = n / Bb, b = n % Bb;
  float o[2], l[2];
#pragma unroll
  for (int r = 0; r < 2; ++r) {
    int brh = (b * Rr + r) * Hh + h;
    int p = invp[(size_t)brh * Tt + t];
    l[r] = lse_s[(size_t)brh * Tt + p];
    o[r] = out_s[((size_t)brh * Tt + p) * Dd + d];
  }
  float mx = fmaxf(l[0], l[1]);
  float w0 = __expf(l[0] - mx), w1 = __expf(l[1] - mx);
  ctx[gid] = (bf16)((w0 * o[0] + w1 * o[1]) / (w0 + w1));
}

// ---------------- host: workspace carving + launch sequence ------------------

extern "C" void kernel_launch(void* const* d_in, const int* in_sizes, int n_in,
                              void* d_out, int out_size, void* d_ws,
                              size_t ws_size, hipStream_t stream) {
  const float* query  = (const float*)d_in[0];
  // d_in[1] key_in unused (share_kq), d_in[5]/[6] wk/bk unused.
  const float* value  = (const float*)d_in[2];
  const float* wq     = (const float*)d_in[3];
  const float* bq     = (const float*)d_in[4];
  const float* wv     = (const float*)d_in[7];
  const float* bv     = (const float*)d_in[8];
  const float* wo     = (const float*)d_in[9];
  const float* bo     = (const float*)d_in[10];
  const float* hash_w = (const float*)d_in[11];
  float* out = (float*)d_out;

  char* ws = (char*)d_ws;
  size_t off = 0;
  auto carve = [&](size_t bytes) -> void* {
    void* p = ws + off;
    off += (bytes + 255) & ~(size_t)255;
    return p;
  };
  float* qf32   = (float*)carve((size_t)Nrows * Ee * 4);
  bf16*  vbf    = (bf16*)carve((size_t)Nrows * Ee * 2);
  bf16*  qbf    = (bf16*)carve((size_t)Nrows * Ee * 2);
  bf16*  kbf    = (bf16*)carve((size_t)Nrows * Ee * 2);
  int*   bucket = (int*)carve((size_t)NBRH * Tt * 4);
  int*   idx    = (int*)carve((size_t)NBRH * Tt * 4);
  int*   invp   = (int*)carve((size_t)NBRH * Tt * 4);
  int*   sbuck  = (int*)carve((size_t)NBRH * Tt * 4);
  bf16*  qsrt   = (bf16*)carve((size_t)NBRH * Tt * Dd * 2);
  bf16*  ksrt   = (bf16*)carve((size_t)NBRH * Tt * Dd * 2);
  bf16*  vsrt   = (bf16*)carve((size_t)NBRH * Tt * Dd * 2);
  float* out_s  = (float*)carve((size_t)NBRH * Tt * Dd * 4);
  float* lse_s  = (float*)carve((size_t)NBRH * Tt * 4);
  bf16*  ctx    = (bf16*)carve((size_t)Nrows * Ee * 2);
  (void)ws_size; (void)in_sizes; (void)n_in; (void)out_size;

  dim3 ggrid(Ee / 128, Nrows / 128);  // (8, 64)
  // q = query @ wq^T + bq  (f32 out, needed for norm + hashing)
  gemm_xwT_kernel<false><<<ggrid, 256, 0, stream>>>(query, wq, bq, qf32,
                                                    nullptr, Nrows, Ee, Ee);
  // v = value @ wv^T + bv  (bf16 out)
  gemm_xwT_kernel<false><<<ggrid, 256, 0, stream>>>(value, wv, bv, nullptr,
                                                    vbf, Nrows, Ee, Ee);
  norm_scale_kernel<<<Nrows, 256, 0, stream>>>(qf32, qbf, kbf);
  hash_kernel<<<(Nrows * Rr * Hh) / 256, 256, 0, stream>>>(qf32, hash_w,
                                                           bucket);
  sort_kernel<<<NBRH, 256, 0, stream>>>(bucket, idx, invp, sbuck);
  gather_kernel<<<(NBRH * Tt * Dd) / 256, 256, 0, stream>>>(
      qbf, kbf, vbf, idx, qsrt, ksrt, vsrt);
  lsh_attn_kernel<<<NBRH * Cc, 256, 0, stream>>>(qsrt, ksrt, vsrt, sbuck,
                                                 out_s, lse_s);
  combine_kernel<<<((size_t)Nrows * Ee) / 256, 256, 0, stream>>>(out_s, lse_s,
                                                                 invp, ctx);
  // out = ctx @ wo^T + bo  (f32 to d_out)
  gemm_xwT_kernel<true><<<ggrid, 256, 0, stream>>>(ctx, wo, bo, out, nullptr,
                                                   Nrows, Ee, Ee);
}
